// Method_24781961298223
// MI455X (gfx1250) — matrix operand order repaired and checked by reading, not compiled, not run on hardware
//
#include <hip/hip_runtime.h>
#include <hip/hip_bf16.h>
#include <math.h>

// ---------------------------------------------------------------------------
// Recurrent gated scan (sLSTM-like) for MI455X / gfx1250, wave32 + WMMA bf16.
//
//   G  = [T*B, 2HD] = X@ih + biases      (one big parallel WMMA GEMM)
//   per step t:  H = Q_t @ hh_w          (WMMA GEMM, 16 blocks, serialized
//                F = sigmoid(G_f + H_f)   via stream-ordered kernel launches)
//                A = tanh  (G_a + H_a)
//                Q_{t+1} = F*Q_t + (sig(zeta)*(1-F)+sig(nu))*A  -> YALL2[t]
//
// Workspace layout (assumed ws_size >= ~328 MB):
//   [0)                G      f32  T*B*2HD          = 256 MB
//   [+256MB)           xbf    bf16 T*B*ID           =  64 MB
//   [+320MB)           ih_t   bf16 2HD x ID  (transposed, [N][K]) = 4 MB
//   [+324MB)           hh_t   bf16 2HD x HD  (transposed, [N][K]) = 4 MB
// ---------------------------------------------------------------------------

typedef __attribute__((ext_vector_type(16))) __bf16 v16bf;
typedef __attribute__((ext_vector_type(8)))  float  v8f;

#define T_STEPS 512
#define BATCH   64
#define ID_DIM  1024
#define HD_DIM  1024
#define N2      2048                 // 2*HD
#define TB      (T_STEPS * BATCH)    // 32768

union FragBF {
    v16bf v;
    uint4 u[2];
};

__device__ __forceinline__ float sigmoidf_(float x) {
    return 1.0f / (1.0f + __expf(-x));
}

// ------------------------------- prep --------------------------------------
__global__ void k_cvt_x(const float* __restrict__ x, __bf16* __restrict__ xbf, int n) {
    int i = blockIdx.x * blockDim.x + threadIdx.x;
    if (i < n) xbf[i] = (__bf16)x[i];
}

// wt[n][k] = (bf16) w[k][n]   (pre-swizzle weights so WMMA B-fragments are
// contiguous 32B-per-lane global_load_b128 pairs; hh_t stays L2-resident)
__global__ void k_transpose_bf(const float* __restrict__ w, __bf16* __restrict__ wt,
                               int rowsK, int colsN) {
    int i = blockIdx.x * blockDim.x + threadIdx.x;
    if (i < rowsK * colsN) {
        int k = i / colsN;
        int n = i % colsN;
        wt[(size_t)n * rowsK + k] = (__bf16)w[(size_t)k * colsN + n];
    }
}

// --------------------- phase 1: G = X@ih + bias ----------------------------
// block = 256 thr (8 waves), block tile M=256 x N=64; wave tile 32 x 64
// (two A-fragments share each B-fragment: 8 WMMAs per 12 x 16B loads).
// grid  = (TB/256, N2/64) = (128, 32)
__global__ __launch_bounds__(256)
void k_input_gemm(const __bf16* __restrict__ xbf,   // [TB][ID]
                  const __bf16* __restrict__ iht,   // [N2][ID]
                  const float*  __restrict__ f_b,
                  const float*  __restrict__ a_b,
                  const float*  __restrict__ PFB,
                  float*        __restrict__ G)     // [TB][N2]
{
    const int lane   = threadIdx.x & 31;
    const int wave   = threadIdx.x >> 5;
    const int laneHi = lane >> 4;
    const int laneLo = lane & 15;

    const int Mg = blockIdx.x * 256 + wave * 32;    // wave covers rows [Mg, Mg+32)
    const int Ng = blockIdx.y * 64;

    v8f acc[2][4] = {};
    const __bf16* aptr0 = xbf + (size_t)(Mg + laneLo) * ID_DIM;       // M-tile 0
    const __bf16* aptr1 = xbf + (size_t)(Mg + 16 + laneLo) * ID_DIM;  // M-tile 1

    for (int kb = 0; kb < ID_DIM; kb += 32) {
        // A 16x32 bf16: elems 0..7 -> K = kb + 8*laneHi + i
        //               elems 8..15 -> K = kb + 16 + 8*laneHi + (i-8)
        FragBF a0, a1;
        a0.u[0] = *(const uint4*)(aptr0 + kb + laneHi * 8);
        a0.u[1] = *(const uint4*)(aptr0 + kb + 16 + laneHi * 8);
        a1.u[0] = *(const uint4*)(aptr1 + kb + laneHi * 8);
        a1.u[1] = *(const uint4*)(aptr1 + kb + 16 + laneHi * 8);
#pragma unroll
        for (int j = 0; j < 4; ++j) {
            // B 32x16 bf16: lane -> N, elems 0..7 -> K = kb + 8*laneHi + i, 8..15 -> kb + 16 + 8*laneHi + (i-8)
            const __bf16* bptr =
                iht + (size_t)(Ng + j * 16 + laneLo) * ID_DIM + kb + laneHi * 8;
            FragBF b;
            b.u[0] = *(const uint4*)(bptr);
            b.u[1] = *(const uint4*)(bptr + 16);
            acc[0][j] = __builtin_amdgcn_wmma_f32_16x16x32_bf16(
                false, a0.v, false, b.v, (short)0, acc[0][j], false, false);
            acc[1][j] = __builtin_amdgcn_wmma_f32_16x16x32_bf16(
                false, a1.v, false, b.v, (short)0, acc[1][j], false, false);
        }
    }

#pragma unroll
    for (int m = 0; m < 2; ++m) {
#pragma unroll
        for (int j = 0; j < 4; ++j) {
            const int gcol = Ng + j * 16 + laneLo;
            const float bias = (gcol < HD_DIM) ? (f_b[gcol] + PFB[gcol])
                                               : a_b[gcol - HD_DIM];
#pragma unroll
            for (int r = 0; r < 8; ++r) {
                const int grow = Mg + m * 16 + laneHi * 8 + r; // D: r -> M=8*laneHi+r
                G[(size_t)grow * N2 + gcol] = acc[m][j][r] + bias;
            }
        }
    }
}

// --------------------- phase 2: one recurrent step -------------------------
// grid = HD/64 = 16 blocks; block = 256 thr (8 waves).
// waves 0-3: F-gate columns [jb, jb+64); waves 4-7: A columns; each wave 16x64.
__global__ __launch_bounds__(256)
void k_step(const float*  __restrict__ Qprev,  // [B][HD] f32 (Q0 or Y[t-1])
            const __bf16* __restrict__ hht,    // [N2][HD] bf16
            const float*  __restrict__ G,      // [TB][N2]
            const float*  __restrict__ zeta,
            const float*  __restrict__ nu,
            float*        __restrict__ Yt,     // [B][HD] -> Q_{t+1}
            int t)
{
    __shared__ float Hs[2][64][64];            // 32 KB: F-half / A-half accums

    const int lane   = threadIdx.x & 31;
    const int wave   = threadIdx.x >> 5;
    const int laneHi = lane >> 4;
    const int laneLo = lane & 15;
    const int half   = wave >> 2;              // 0 = F columns, 1 = A columns
    const int mrow   = (wave & 3) * 16;
    const int jb     = blockIdx.x * 64;

    v8f acc[4] = {};
    const float* qrow = Qprev + (size_t)(mrow + laneLo) * HD_DIM;

    for (int kb = 0; kb < HD_DIM; kb += 32) {
        // build bf16 A-fragment from f32 Q (convert on load)
        const float* g0 = qrow + kb + laneHi * 8;
        const float* g1 = qrow + kb + 16 + laneHi * 8;
        const float4 f0 = *(const float4*)(g0);
        const float4 f1 = *(const float4*)(g0 + 4);
        const float4 f2 = *(const float4*)(g1);
        const float4 f3 = *(const float4*)(g1 + 4);
        v16bf av = { (__bf16)f0.x, (__bf16)f0.y, (__bf16)f0.z, (__bf16)f0.w,
                     (__bf16)f1.x, (__bf16)f1.y, (__bf16)f1.z, (__bf16)f1.w,
                     (__bf16)f2.x, (__bf16)f2.y, (__bf16)f2.z, (__bf16)f2.w,
                     (__bf16)f3.x, (__bf16)f3.y, (__bf16)f3.z, (__bf16)f3.w };
#pragma unroll
        for (int j = 0; j < 4; ++j) {
            const int n = half * HD_DIM + jb + j * 16 + laneLo;
            const __bf16* bptr = hht + (size_t)n * HD_DIM + kb + laneHi * 8;
            FragBF b;
            b.u[0] = *(const uint4*)(bptr);
            b.u[1] = *(const uint4*)(bptr + 16);
            acc[j] = __builtin_amdgcn_wmma_f32_16x16x32_bf16(
                false, av, false, b.v, (short)0, acc[j], false, false);
        }
    }

    // exchange F/A partial sums through LDS
#pragma unroll
    for (int j = 0; j < 4; ++j)
#pragma unroll
        for (int r = 0; r < 8; ++r)
            Hs[half][mrow + laneHi * 8 + r][j * 16 + laneLo] = acc[j][r];
    __syncthreads();

    // fused gating pointwise: 4096 elements over 256 threads
    const float sigz = sigmoidf_(zeta[0]);
    const float sign_ = sigmoidf_(nu[0]);
    const int row = threadIdx.x >> 2;          // 0..63 (batch)
    const int cb  = (threadIdx.x & 3) * 16;
    const float* Grow = G + (size_t)(t * BATCH + row) * N2;
#pragma unroll
    for (int c = 0; c < 16; ++c) {
        const int col = cb + c;
        const int j   = jb + col;
        const float F = sigmoidf_(Grow[j] + Hs[0][row][col]);
        const float A = tanhf(Grow[HD_DIM + j] + Hs[1][row][col]);
        const float I = sigz * (1.0f - F) + sign_;
        Yt[(size_t)row * HD_DIM + j] =
            F * Qprev[(size_t)row * HD_DIM + j] + I * A;
    }
}

// ------------------------------- tails -------------------------------------
__global__ void k_final(const float* __restrict__ Ylast,
                        float* __restrict__ Qout, float* __restrict__ Sout, int n) {
    int i = blockIdx.x * blockDim.x + threadIdx.x;
    if (i < n) { float v = Ylast[i]; Qout[i] = v; Sout[i] = v; }
}

// ---------------------------------------------------------------------------
extern "C" void kernel_launch(void* const* d_in, const int* in_sizes, int n_in,
                              void* d_out, int out_size, void* d_ws, size_t ws_size,
                              hipStream_t stream)
{
    (void)in_sizes; (void)n_in; (void)out_size; (void)ws_size;

    const float* x    = (const float*)d_in[0];
    const float* Q0   = (const float*)d_in[1];
    /* d_in[2] = S0, unused by the reference computation */
    const float* ih   = (const float*)d_in[3];
    const float* hh_w = (const float*)d_in[4];
    const float* f_b  = (const float*)d_in[5];
    const float* a_b  = (const float*)d_in[6];
    const float* PFB  = (const float*)d_in[7];
    const float* zeta = (const float*)d_in[8];
    const float* nu   = (const float*)d_in[9];

    float* Y    = (float*)d_out;                        // [T][B][HD]
    float* Qout = Y + (size_t)T_STEPS * BATCH * HD_DIM; // [B][HD]
    float* Sout = Qout + (size_t)BATCH * HD_DIM;        // [B][HD]

    char* ws = (char*)d_ws;
    float*  G   = (float*)ws;
    __bf16* xbf = (__bf16*)(ws + (size_t)TB * N2 * 4);
    __bf16* iht = (__bf16*)(ws + (size_t)TB * N2 * 4 + (size_t)TB * ID_DIM * 2);
    __bf16* hht = iht + (size_t)N2 * ID_DIM;

    // --- prep: bf16 conversion + weight pre-swizzle -------------------------
    {
        const int nx = TB * ID_DIM;
        k_cvt_x<<<(nx + 255) / 256, 256, 0, stream>>>(x, xbf, nx);
        const int nw = ID_DIM * N2;
        k_transpose_bf<<<(nw + 255) / 256, 256, 0, stream>>>(ih,   iht, ID_DIM, N2);
        k_transpose_bf<<<(nw + 255) / 256, 256, 0, stream>>>(hh_w, hht, HD_DIM, N2);
    }

    // --- phase 1: all input projections at once -----------------------------
    {
        dim3 grid(TB / 256, N2 / 64);
        k_input_gemm<<<grid, 256, 0, stream>>>(xbf, iht, f_b, a_b, PFB, G);
    }

    // --- phase 2: sequential scan; Y[t] doubles as Q state ------------------
    for (int t = 0; t < T_STEPS; ++t) {
        const float* Qprev = (t == 0) ? Q0 : (Y + (size_t)(t - 1) * BATCH * HD_DIM);
        k_step<<<HD_DIM / 64, 256, 0, stream>>>(
            Qprev, hht, G, zeta, nu, Y + (size_t)t * BATCH * HD_DIM, t);
    }

    // --- tails: Q_out / S_out -----------------------------------------------
    k_final<<<(BATCH * HD_DIM + 255) / 256, 256, 0, stream>>>(
        Y + (size_t)(T_STEPS - 1) * BATCH * HD_DIM, Qout, Sout, BATCH * HD_DIM);
}
